// GAT_30932354465908
// MI455X (gfx1250) — matrix operand order repaired and checked by reading, not compiled, not run on hardware
//
#include <hip/hip_runtime.h>
#include <hip/hip_bf16.h>
#include <stdint.h>

// ---------------------------------------------------------------------------
// Problem constants (match reference)
// ---------------------------------------------------------------------------
#define N_NODES 50000
#define N_EDGES 500000
#define IN_DIM  512
#define HID     256
#define OUT_DIM 64

typedef __attribute__((ext_vector_type(4)))  __bf16 v4bf;
typedef __attribute__((ext_vector_type(8)))  __bf16 v8bf;
typedef __attribute__((ext_vector_type(16))) __bf16 v16bf;
typedef __attribute__((ext_vector_type(8)))  float  v8f;

static __device__ __forceinline__ __bf16 f2bf(float f) {
  union { float f; unsigned u; } v; v.f = f;
  unsigned r = (v.u + 0x7FFFu + ((v.u >> 16) & 1u)) >> 16;  // RNE
  union { unsigned short s; __bf16 b; } o; o.s = (unsigned short)r;
  return o.b;
}

// ---------------------------------------------------------------------------
// Streaming f32 -> bf16 convert (row-major layout preserved), float4 in,
// 8-byte vector out. n4 = element count / 4.
// ---------------------------------------------------------------------------
__global__ __launch_bounds__(256)
void cvt_bf16_k(const float* __restrict__ src, __bf16* __restrict__ dst,
                long long n4) {
  long long i = (long long)blockIdx.x * blockDim.x + threadIdx.x;
  if (i >= n4) return;
  float4 v = ((const float4*)src)[i];
  v4bf o;
  o[0] = f2bf(v.x); o[1] = f2bf(v.y); o[2] = f2bf(v.z); o[3] = f2bf(v.w);
  ((v4bf*)dst)[i] = o;
}

// ---------------------------------------------------------------------------
// Pack weight matrix into WMMA B-fragment layout, bf16:
//   fragment f = kb*(N/16)+ct ; lane l ; element e
//   value = B[(kb*32 + 16*(l>=16) + e) * bsk + (ct*16 + l%16) * bsn]
//   stored at Bp[(f*32 + l)*16 + e]  -> 32 contiguous bytes per lane
// ---------------------------------------------------------------------------
__global__ __launch_bounds__(256)
void pack_b_frag(const float* __restrict__ B, __bf16* __restrict__ Bp,
                 int K, int N, int bsk, int bsn) {
  int tid = blockIdx.x * blockDim.x + threadIdx.x;
  int total = (K >> 5) * (N >> 4) * 32;
  if (tid >= total) return;
  int lane = tid & 31;
  int f = tid >> 5;
  int nct = N >> 4;
  int ct = f % nct;
  int kb = f / nct;
  int kbase = kb * 32 + ((lane >> 4) << 4);
  int n = ct * 16 + (lane & 15);
  v16bf o;
#pragma unroll
  for (int e = 0; e < 16; ++e)
    o[e] = f2bf(B[(size_t)(kbase + e) * bsk + (size_t)n * bsn]);
  *(v16bf*)(Bp + (size_t)tid * 16) = o;
}

// ---------------------------------------------------------------------------
// bf16 WMMA GEMM:  C[M,N] (f32) = A[M,K] (bf16 row-major) x Bp (packed frags)
// One wave = 16x64 output strip (4 WMMA tiles). Block = 4 waves.
// A fragment: two 16B v8bf loads per lane; B fragment: one 32B v16bf load.
// ---------------------------------------------------------------------------
__global__ __launch_bounds__(128)
void gemm_bf16_wmma(const __bf16* __restrict__ A, const __bf16* __restrict__ Bp,
                    float* __restrict__ C, int M, int N, int K) {
  const int wave = threadIdx.x >> 5;
  const int lane = threadIdx.x & 31;
  const int half = lane >> 4;
  const int l16  = lane & 15;
  const int rowTile = blockIdx.x * 4 + wave;
  if (rowTile * 16 >= M) return;
  const int row0 = rowTile * 16;
  const int ct0  = blockIdx.y * 4;   // first of 4 col tiles
  const int nct  = N >> 4;

  v8f acc[4] = {};

  const __bf16* arow = A + (size_t)(row0 + l16) * K + half * 16;
  const __bf16* bp = Bp + ((size_t)ct0 * 32 + lane) * 16;
  const size_t bStrideK = (size_t)nct * 32 * 16;   // per kb
  const size_t bStrideT = (size_t)32 * 16;         // per col tile

  for (int k0 = 0; k0 < K; k0 += 32) {
    v8bf lo = *(const v8bf*)(arow + k0);        // K = half*16 + {0..7}
    v8bf hi = *(const v8bf*)(arow + k0 + 8);    // K = half*16 + {8..15}
    v16bf a = __builtin_shufflevector(lo, hi, 0, 1, 2, 3, 4, 5, 6, 7,
                                      8, 9, 10, 11, 12, 13, 14, 15);
#pragma unroll
    for (int t = 0; t < 4; ++t) {
      v16bf b = *(const v16bf*)(bp + (size_t)t * bStrideT);
      acc[t] = __builtin_amdgcn_wmma_f32_16x16x32_bf16(
          false, a, false, b, (short)0, acc[t], false, false);
    }
    bp += bStrideK;
  }

  // C/D layout: VGPR j -> row = j + 8*half, col = lane%16
#pragma unroll
  for (int t = 0; t < 4; ++t) {
    float* cp = C + (size_t)(row0 + half * 8) * N + ((ct0 + t) * 16 + l16);
#pragma unroll
    for (int j = 0; j < 8; ++j) cp[(size_t)j * N] = acc[t][j];
  }
}

// ---------------------------------------------------------------------------
// Node attention scores: a_src[n] = h[n]·att_src, a_dst[n] = h[n]·att_dst
// ---------------------------------------------------------------------------
__global__ __launch_bounds__(256)
void att_dot(const float* __restrict__ h, const float* __restrict__ atts,
             const float* __restrict__ attd, float* __restrict__ as,
             float* __restrict__ ad, int n, int C) {
  int w = (blockIdx.x * blockDim.x + threadIdx.x) >> 5;
  int lane = threadIdx.x & 31;
  if (w >= n) return;
  const float* row = h + (size_t)w * C;
  float s0 = 0.f, s1 = 0.f;
  for (int c = lane; c < C; c += 32) {
    float v = row[c];
    s0 += v * atts[c];
    s1 += v * attd[c];
  }
#pragma unroll
  for (int off = 16; off > 0; off >>= 1) {
    s0 += __shfl_down(s0, off, 32);
    s1 += __shfl_down(s1, off, 32);
  }
  if (lane == 0) { as[w] = s0; ad[w] = s1; }
}

// Edge logits: a = sigmoid(as[src]+ad[dst]); segment-max via uint atomicMax
// (logits in (0,1) so uint order == float order; init 0 matches reference).
__global__ __launch_bounds__(256)
void edge_logits(const int* __restrict__ src, const int* __restrict__ dst,
                 const float* __restrict__ as, const float* __restrict__ ad,
                 float* __restrict__ w, unsigned* __restrict__ mx, int E) {
  int e = blockIdx.x * blockDim.x + threadIdx.x;
  if (e >= E) return;
  float a = as[src[e]] + ad[dst[e]];
  a = 1.f / (1.f + __expf(-a));
  w[e] = a;
  atomicMax(mx + dst[e], __float_as_uint(a));
}

__global__ __launch_bounds__(256)
void edge_exp(float* __restrict__ w, const unsigned* __restrict__ mx,
              const int* __restrict__ dst, float* __restrict__ denom, int E) {
  int e = blockIdx.x * blockDim.x + threadIdx.x;
  if (e >= E) return;
  int d = dst[e];
  float ex = __expf(w[e] - __uint_as_float(mx[d]));
  w[e] = ex;
  atomicAdd(denom + d, ex);
}

// Weighted scatter-add: out[dst] += h[src] * (w / (denom[dst]+1e-16)).
// One wave per edge, float4 gathers, f32 atomics resolve in 192MB L2.
__global__ __launch_bounds__(256)
void scatter_k(const float* __restrict__ h, const float* __restrict__ w,
               const float* __restrict__ denom, const int* __restrict__ src,
               const int* __restrict__ dst, float* __restrict__ out,
               int E, int C) {
  int e = (blockIdx.x * blockDim.x + threadIdx.x) >> 5;
  int lane = threadIdx.x & 31;
  if (e >= E) return;
  int s = src[e], d = dst[e];
  float wt = w[e] / (denom[d] + 1e-16f);
  const float4* hs = (const float4*)(h + (size_t)s * C);
  float* od = out + (size_t)d * C;
  for (int c4 = lane; c4 < (C >> 2); c4 += 32) {
    float4 v = hs[c4];
    atomicAdd(od + 4 * c4 + 0, v.x * wt);
    atomicAdd(od + 4 * c4 + 1, v.y * wt);
    atomicAdd(od + 4 * c4 + 2, v.z * wt);
    atomicAdd(od + 4 * c4 + 3, v.w * wt);
  }
}

__global__ __launch_bounds__(256)
void elu_k(float* __restrict__ p, long long n) {
  long long i = (long long)blockIdx.x * blockDim.x + threadIdx.x;
  if (i >= n) return;
  float x = p[i];
  p[i] = x > 0.f ? x : (__expf(x) - 1.f);
}

__global__ __launch_bounds__(256)
void zero_k(float* __restrict__ p, long long n) {
  long long i = (long long)blockIdx.x * blockDim.x + threadIdx.x;
  if (i < n) p[i] = 0.f;
}

// ---------------------------------------------------------------------------
extern "C" void kernel_launch(void* const* d_in, const int* in_sizes, int n_in,
                              void* d_out, int out_size, void* d_ws, size_t ws_size,
                              hipStream_t stream) {
  (void)in_sizes; (void)n_in; (void)out_size; (void)ws_size;
  const float* features = (const float*)d_in[0];   // [N, 512]
  const float* W1       = (const float*)d_in[1];   // [256, 512]
  const float* W2       = (const float*)d_in[2];   // [64, 256]
  const float* att_src  = (const float*)d_in[3];   // [256]
  const float* att_dst  = (const float*)d_in[4];   // [256]
  const int*   srcI     = (const int*)d_in[5];     // edge_index[0]
  const int*   dstI     = srcI + N_EDGES;          // edge_index[1]

  float* h2 = (float*)d_out;                              // [N, 64]
  float* h4 = (float*)d_out + (size_t)N_NODES * OUT_DIM;  // [N, 512]

  // h4 output region (25.6M f32 = 102.4MB) doubles as scratch until the
  // final GEMM: first half = bufA (h1p/h3p f32), second half = bf16 features.
  float*  bufA    = h4;                                        // [N,256] f32
  __bf16* AbfFeat = (__bf16*)(h4 + (size_t)N_NODES * HID);     // [N,512] bf16

  // Workspace layout
  float* bufB   = (float*)d_ws;                    // [N,256] f32 aggregates
  float* w_edge = bufB + (size_t)N_NODES * HID;    // [E]
  float* mxf    = w_edge + N_EDGES;                // [N] (uint bits)
  float* denom  = mxf + N_NODES;                   // [N]
  float* aS     = denom + N_NODES;                 // [N]
  float* aD     = aS + N_NODES;                    // [N]
  __bf16* AbfSmall = (__bf16*)(aD + N_NODES);      // [N,256] bf16 (reused)
  __bf16* Bp1 = AbfSmall + (size_t)N_NODES * HID;  // W1^T packed  512x256
  __bf16* Bp2 = Bp1 + (size_t)IN_DIM * HID;        // W2^T packed  256x64
  __bf16* Bp3 = Bp2 + (size_t)HID * OUT_DIM;       // W2   packed   64x256
  __bf16* Bp4 = Bp3 + (size_t)OUT_DIM * HID;       // W1   packed  256x512

  const int ROWT = (N_NODES / 16 + 3) / 4;  // 782 blocks of 4 row-tiles
  const long long NH = (long long)N_NODES * HID;          // 12.8M
  const long long NF = (long long)N_NODES * IN_DIM;       // 25.6M
  const long long NO = (long long)N_NODES * OUT_DIM;      // 3.2M

  // segment max/denominator init (contiguous)
  zero_k<<<(2 * N_NODES + 255) / 256, 256, 0, stream>>>(mxf, 2 * N_NODES);

  // pack all four weight views (tiny, L2-resident afterwards)
  pack_b_frag<<<(IN_DIM / 32 * HID / 16 * 32 + 255) / 256, 256, 0, stream>>>(
      W1, Bp1, IN_DIM, HID, 1, IN_DIM);          // B[k][n] = W1[n*512+k]
  pack_b_frag<<<(HID / 32 * OUT_DIM / 16 * 32 + 255) / 256, 256, 0, stream>>>(
      W2, Bp2, HID, OUT_DIM, 1, HID);            // B[k][n] = W2[n*256+k]
  pack_b_frag<<<(OUT_DIM / 32 * HID / 16 * 32 + 255) / 256, 256, 0, stream>>>(
      W2, Bp3, OUT_DIM, HID, HID, 1);            // B[k][n] = W2[k*256+n]
  pack_b_frag<<<(HID / 32 * IN_DIM / 16 * 32 + 255) / 256, 256, 0, stream>>>(
      W1, Bp4, HID, IN_DIM, IN_DIM, 1);          // B[k][n] = W1[k*512+n]

  // conv1 linear: h1p = features @ W1.T
  cvt_bf16_k<<<(int)((NF / 4 + 255) / 256), 256, 0, stream>>>(
      features, AbfFeat, NF / 4);
  gemm_bf16_wmma<<<dim3(ROWT, HID / 64), 128, 0, stream>>>(
      AbfFeat, Bp1, bufA, N_NODES, HID, IN_DIM);

  // tied node attention scores + softmax weights (computed once)
  att_dot<<<(N_NODES * 32 + 255) / 256, 256, 0, stream>>>(
      bufA, att_src, att_dst, aS, aD, N_NODES, HID);
  edge_logits<<<(N_EDGES + 255) / 256, 256, 0, stream>>>(
      srcI, dstI, aS, aD, w_edge, (unsigned*)mxf, N_EDGES);
  edge_exp<<<(N_EDGES + 255) / 256, 256, 0, stream>>>(
      w_edge, (const unsigned*)mxf, dstI, denom, N_EDGES);

  // propagate 1: h1 = elu(scatter(h1p * w))
  zero_k<<<(int)((NH + 255) / 256), 256, 0, stream>>>(bufB, NH);
  scatter_k<<<(N_EDGES * 32 + 255) / 256, 256, 0, stream>>>(
      bufA, w_edge, denom, srcI, dstI, bufB, N_EDGES, HID);
  elu_k<<<(int)((NH + 255) / 256), 256, 0, stream>>>(bufB, NH);

  // conv2: h2 = h1 @ W2.T
  cvt_bf16_k<<<(int)((NH / 4 + 255) / 256), 256, 0, stream>>>(
      bufB, AbfSmall, NH / 4);
  gemm_bf16_wmma<<<dim3(ROWT, OUT_DIM / 64), 128, 0, stream>>>(
      AbfSmall, Bp2, h2, N_NODES, OUT_DIM, HID);

  // conv3 linear: h3p = h2 @ W2
  cvt_bf16_k<<<(int)((NO / 4 + 255) / 256), 256, 0, stream>>>(
      h2, AbfSmall, NO / 4);
  gemm_bf16_wmma<<<dim3(ROWT, HID / 64), 128, 0, stream>>>(
      AbfSmall, Bp3, bufA, N_NODES, HID, OUT_DIM);

  // propagate 2 (tied weights): h3 = elu(scatter(h3p * w))
  zero_k<<<(int)((NH + 255) / 256), 256, 0, stream>>>(bufB, NH);
  scatter_k<<<(N_EDGES * 32 + 255) / 256, 256, 0, stream>>>(
      bufA, w_edge, denom, srcI, dstI, bufB, N_EDGES, HID);
  elu_k<<<(int)((NH + 255) / 256), 256, 0, stream>>>(bufB, NH);

  // conv4: h4 = h3 @ W1  (bufA/AbfFeat region dead; safe to overwrite)
  cvt_bf16_k<<<(int)((NH / 4 + 255) / 256), 256, 0, stream>>>(
      bufB, AbfSmall, NH / 4);
  gemm_bf16_wmma<<<dim3(ROWT, IN_DIM / 64), 128, 0, stream>>>(
      AbfSmall, Bp4, h4, N_NODES, IN_DIM, HID);
}
